// Attention_17059610099953
// MI455X (gfx1250) — compile-verified, matrix operand order checked by reading, not run on hardware
//
#include <hip/hip_runtime.h>
#include <hip/hip_bf16.h>

typedef __attribute__((ext_vector_type(16))) __bf16 v16bf;
typedef __attribute__((ext_vector_type(8)))  __bf16 v8bf;
typedef __attribute__((ext_vector_type(8)))  float  v8f;
typedef __attribute__((ext_vector_type(4)))  float  v4f;

// ---- constants for this problem instance ----
constexpr int NWIN = 50;      // B(2) * 5*5 windows
constexpr int TOK  = 400;     // t*ws*ws
constexpr int C    = 384;
constexpr int NH   = 12;
constexpr int HD   = 32;
constexpr int C3   = 1152;

static __device__ __forceinline__ v8f wmma_bf16(v16bf a, v16bf b, v8f c) {
  return __builtin_amdgcn_wmma_f32_16x16x32_bf16(false, a, false, b, (short)0, c,
                                                 false, false);
}

static __device__ __forceinline__ v16bf cat16(v8bf a, v8bf b) {
  return __builtin_shufflevector(a, b, 0,1,2,3,4,5,6,7,8,9,10,11,12,13,14,15);
}

static __device__ __forceinline__ v8bf ld8(const __bf16* p) {
  return *(const v8bf*)p;                 // 16B contiguous bf16 chunk
}

static __device__ __forceinline__ v8f zero8() {
  v8f z = {0.f,0.f,0.f,0.f,0.f,0.f,0.f,0.f};
  return z;
}

// Async copy of one 16B chunk: global -> LDS (gfx1250 ASYNCcnt path).
static __device__ __forceinline__ void async_g2l_b128(void* lds_dst,
                                                      const void* gaddr) {
  unsigned ldsoff = (unsigned)(size_t)lds_dst;       // low 32 bits = LDS addr
  asm volatile("global_load_async_to_lds_b128 %0, %1, off"
               :: "v"(ldsoff), "v"((unsigned long long)(size_t)gaddr)
               : "memory");
}
static __device__ __forceinline__ void wait_async0() {
  asm volatile("s_wait_asynccnt 0" ::: "memory");
}

// ------------------------------------------------------------------
// Kernel 0a: window partition + zero pad + fp32 -> bf16, once.
// ------------------------------------------------------------------
__global__ __launch_bounds__(256) void gather_win(const float* __restrict__ x,
                                                  __bf16* __restrict__ xw) {
  int idx = blockIdx.x * 256 + threadIdx.x;          // tasks: 50*400*96
  if (idx >= NWIN * TOK * (C / 4)) return;
  int c4 = idx % (C / 4);
  int rs = idx / (C / 4);
  int s  = rs % TOK;
  int w  = rs / TOK;
  int b = w / 25, wr = w % 25, hgi = wr / 5, wgi = wr % 5;
  int tt = s / 100, r2 = s % 100, hh = r2 / 10, ww = r2 % 10;
  int Hc = hgi * 10 + hh, Wc = wgi * 10 + ww;
  v4f val = {0.f, 0.f, 0.f, 0.f};
  if (Hc < 44 && Wc < 44) {
    const float* p = x + (((size_t)(b * 4 + tt) * 1936 + Hc * 44 + Wc) * C + c4 * 4);
    val = *(const v4f*)p;
  }
  __bf16* o = xw + ((size_t)w * TOK + s) * C + c4 * 4;
  o[0] = (__bf16)val[0]; o[1] = (__bf16)val[1];
  o[2] = (__bf16)val[2]; o[3] = (__bf16)val[3];
}

// ------------------------------------------------------------------
// Kernel 0b: weight fp32 -> bf16 (once; keeps cvts out of GEMM loops).
// ------------------------------------------------------------------
__global__ __launch_bounds__(256) void cvt_weights(const float* __restrict__ src,
                                                   __bf16* __restrict__ dst,
                                                   int n4) {
  int i = blockIdx.x * 256 + threadIdx.x;
  if (i >= n4) return;
  v4f v = *(const v4f*)(src + (size_t)i * 4);
  __bf16* o = dst + (size_t)i * 4;
  o[0] = (__bf16)v[0]; o[1] = (__bf16)v[1];
  o[2] = (__bf16)v[2]; o[3] = (__bf16)v[3];
}

// ------------------------------------------------------------------
// Kernel 1: QKV GEMM.  A-panel (80x384 bf16 = 60 KB) staged in LDS via
// async global->LDS; waves read batched A-fragments (one ds clause per
// k-step), B-fragment double-buffered from global.
// ------------------------------------------------------------------
__global__ __launch_bounds__(256) void qkv_gemm(const __bf16* __restrict__ xw,
                                                const __bf16* __restrict__ wb,
                                                __bf16* __restrict__ q,
                                                __bf16* __restrict__ kk,
                                                __bf16* __restrict__ vt) {
  __shared__ __bf16 As[80 * C];              // 61,440 bytes
  int w    = blockIdx.x / 5;
  int mb   = blockIdx.x % 5;                 // 80-row block
  int wave = threadIdx.x >> 5, lane = threadIdx.x & 31;
  int half = lane >> 4, colL = lane & 15, off = half * 8;

  {  // async stage A panel: 80*384 bf16 = 3840 chunks of 16B
    const __bf16* Ab = xw + ((size_t)w * TOK + mb * 80) * C;
    for (int i = threadIdx.x; i < 80 * C / 8; i += 256)   // 15 iters exactly
      async_g2l_b128(&As[i * 8], Ab + (size_t)i * 8);
    wait_async0();
  }
  __syncthreads();

  for (int np = 0; np < C3 / 128; ++np) {    // 9 column passes
    int n = np * 128 + wave * 16 + colL;
    const __bf16* Brow = wb + (size_t)n * C;
    v8f acc[5];
    #pragma unroll
    for (int i = 0; i < 5; ++i) acc[i] = zero8();

    v16bf bfrag = cat16(ld8(Brow + off), ld8(Brow + 16 + off));
    for (int ks = 0; ks < C / 32; ++ks) {    // 12 K steps
      int k0 = ks * 32;
      v16bf bnext = bfrag;
      if (ks + 1 < C / 32) {                 // prefetch next B fragment
        int k1 = k0 + 32;
        bnext = cat16(ld8(Brow + k1 + off), ld8(Brow + k1 + 16 + off));
      }
      v16bf af[5];                           // batch ds loads -> one wait
      #pragma unroll
      for (int mt = 0; mt < 5; ++mt) {
        const __bf16* Ar = &As[(mt * 16 + colL) * C + k0 + off];
        af[mt] = cat16(ld8(Ar), ld8(Ar + 16));
      }
      #pragma unroll
      for (int mt = 0; mt < 5; ++mt)
        acc[mt] = wmma_bf16(af[mt], bfrag, acc[mt]);
      bfrag = bnext;
    }

    int e = n / C, r = n % C, h = r >> 5, d = r & 31;   // e uniform per wave
    #pragma unroll
    for (int mt = 0; mt < 5; ++mt) {
      #pragma unroll
      for (int j = 0; j < 8; ++j) {
        int s = mb * 80 + mt * 16 + j + half * 8;
        float v = acc[mt][j];
        size_t hb = ((size_t)w * NH + h);
        if (e == 0)      q [(hb * TOK + s) * HD + d] = (__bf16)v;
        else if (e == 1) kk[(hb * TOK + s) * HD + d] = (__bf16)v;
        else             vt[(hb * HD + d) * TOK + s] = (__bf16)v;
      }
    }
  }
}

// ------------------------------------------------------------------
// Kernel 2: per (window, head) flash-style attention, 400q x 400k.
// V fragments loaded before the softmax VALU block so global latency
// overlaps the exp/shuffle work.
// ------------------------------------------------------------------
__global__ __launch_bounds__(256) void attn_kernel(const __bf16* __restrict__ q,
                                                   const __bf16* __restrict__ kk,
                                                   const __bf16* __restrict__ vt,
                                                   __bf16* __restrict__ ow) {
  __shared__ __bf16 pLds[8][16][32];
  int w = blockIdx.x / NH, h = blockIdx.x % NH;
  int wave = threadIdx.x >> 5, lane = threadIdx.x & 31;
  int half = lane >> 4, colL = lane & 15, off = half * 8;
  const float scale = 0.17677669529663688f;          // hd^-0.5

  size_t hb = (size_t)w * NH + h;
  const __bf16* Qb = q  + hb * TOK * HD;
  const __bf16* Kb = kk + hb * TOK * HD;
  const __bf16* Vb = vt + hb * HD * TOK;

  for (int mt = wave; mt < TOK / 16; mt += 8) {      // 25 query tiles
    const __bf16* Qr = Qb + (size_t)(mt * 16 + colL) * HD + off;
    v16bf qf = cat16(ld8(Qr), ld8(Qr + 16));

    v8f o0 = zero8(), o1 = zero8();
    float rmax[8], rsum[8];
    #pragma unroll
    for (int j = 0; j < 8; ++j) { rmax[j] = -1e30f; rsum[j] = 0.f; }

    for (int p = 0; p < 13; ++p) {                   // key tiles in pairs
      int kt0 = 2 * p, kt1 = 2 * p + 1;
      bool has2 = (kt1 < TOK / 16);

      // K fragments for both tiles up front
      const __bf16* Kr0 = Kb + (size_t)(kt0 * 16 + colL) * HD + off;
      v16bf kf0 = cat16(ld8(Kr0), ld8(Kr0 + 16));
      v16bf kf1 = kf0;
      if (has2) {
        const __bf16* Kr1 = Kb + (size_t)(kt1 * 16 + colL) * HD + off;
        kf1 = cat16(ld8(Kr1), ld8(Kr1 + 16));
      }
      // V fragments early: latency hides under softmax VALU below
      int keyBase = kt0 * 16;
      const __bf16* Vr0 = Vb + (size_t)(colL)      * TOK + keyBase + off;
      const __bf16* Vr1 = Vb + (size_t)(16 + colL) * TOK + keyBase + off;
      v8bf v0lo = ld8(Vr0), v0hi = has2 ? ld8(Vr0 + 16) : v0lo;  // dup * P==0
      v8bf v1lo = ld8(Vr1), v1hi = has2 ? ld8(Vr1 + 16) : v1lo;

      v8f s0 = wmma_bf16(qf, kf0, zero8());
      v8f s1;
      if (has2) {
        s1 = wmma_bf16(qf, kf1, zero8());
      } else {
        #pragma unroll
        for (int j = 0; j < 8; ++j) s1[j] = -3e30f;  // exp -> 0
      }

      #pragma unroll
      for (int j = 0; j < 8; ++j) {
        float a = s0[j] * scale, b = s1[j] * scale;
        float m = fmaxf(a, b);
        m = fmaxf(m, __shfl_xor(m, 1));
        m = fmaxf(m, __shfl_xor(m, 2));
        m = fmaxf(m, __shfl_xor(m, 4));
        m = fmaxf(m, __shfl_xor(m, 8));              // stays in 16-lane half
        float nm   = fmaxf(rmax[j], m);
        float corr = __expf(rmax[j] - nm);
        float p0   = __expf(a - nm);
        float p1   = __expf(b - nm);
        float ps   = p0 + p1;
        ps += __shfl_xor(ps, 1); ps += __shfl_xor(ps, 2);
        ps += __shfl_xor(ps, 4); ps += __shfl_xor(ps, 8);
        rsum[j] = rsum[j] * corr + ps;
        rmax[j] = nm;
        o0[j] *= corr; o1[j] *= corr;
        int mrow = j + half * 8;
        pLds[wave][mrow][colL]      = (__bf16)p0;
        pLds[wave][mrow][16 + colL] = (__bf16)p1;
      }
      asm volatile("s_wait_dscnt 0" ::: "memory");   // intra-wave LDS transpose

      const __bf16* Pr = &pLds[wave][colL][0] + off;
      v16bf pf = cat16(ld8(Pr), ld8(Pr + 16));

      o0 = wmma_bf16(pf, cat16(v0lo, v0hi), o0);
      o1 = wmma_bf16(pf, cat16(v1lo, v1hi), o1);
    }

    #pragma unroll
    for (int j = 0; j < 8; ++j) {
      int s = mt * 16 + j + half * 8;
      float inv = 1.0f / rsum[j];
      size_t base = ((size_t)w * TOK + s) * C + h * HD;
      ow[base + colL]      = (__bf16)(o0[j] * inv);
      ow[base + 16 + colL] = (__bf16)(o1[j] * inv);
    }
  }
}

// ------------------------------------------------------------------
// Kernel 3: projection GEMM + bias, scatter to unpadded output (fp32).
// Same staging / batching structure as qkv_gemm.
// ------------------------------------------------------------------
__global__ __launch_bounds__(256) void proj_gemm(const __bf16* __restrict__ ow,
                                                 const __bf16* __restrict__ wb,
                                                 const float*  __restrict__ proj_b,
                                                 float* __restrict__ out) {
  __shared__ __bf16 As[80 * C];
  int w  = blockIdx.x / 5;
  int mb = blockIdx.x % 5;
  int wave = threadIdx.x >> 5, lane = threadIdx.x & 31;
  int half = lane >> 4, colL = lane & 15, off = half * 8;
  int b = w / 25, wr = w % 25, hgi = wr / 5, wgi = wr % 5;

  {
    const __bf16* Ab = ow + ((size_t)w * TOK + mb * 80) * C;
    for (int i = threadIdx.x; i < 80 * C / 8; i += 256)
      async_g2l_b128(&As[i * 8], Ab + (size_t)i * 8);
    wait_async0();
  }
  __syncthreads();

  for (int np = 0; np < C / 128; ++np) {             // 3 column passes
    int n = np * 128 + wave * 16 + colL;
    const __bf16* Brow = wb + (size_t)n * C;
    v8f acc[5];
    #pragma unroll
    for (int i = 0; i < 5; ++i) acc[i] = zero8();

    v16bf bfrag = cat16(ld8(Brow + off), ld8(Brow + 16 + off));
    for (int ks = 0; ks < C / 32; ++ks) {
      int k0 = ks * 32;
      v16bf bnext = bfrag;
      if (ks + 1 < C / 32) {
        int k1 = k0 + 32;
        bnext = cat16(ld8(Brow + k1 + off), ld8(Brow + k1 + 16 + off));
      }
      v16bf af[5];
      #pragma unroll
      for (int mt = 0; mt < 5; ++mt) {
        const __bf16* Ar = &As[(mt * 16 + colL) * C + k0 + off];
        af[mt] = cat16(ld8(Ar), ld8(Ar + 16));
      }
      #pragma unroll
      for (int mt = 0; mt < 5; ++mt)
        acc[mt] = wmma_bf16(af[mt], bfrag, acc[mt]);
      bfrag = bnext;
    }

    float bias = proj_b[n];
    #pragma unroll
    for (int mt = 0; mt < 5; ++mt) {
      #pragma unroll
      for (int j = 0; j < 8; ++j) {
        int s = mb * 80 + mt * 16 + j + half * 8;
        int tt = s / 100, r2 = s % 100, hh = r2 / 10, ww = r2 % 10;
        int Hc = hgi * 10 + hh, Wc = wgi * 10 + ww;
        if (Hc < 44 && Wc < 44) {
          out[((size_t)(b * 4 + tt) * 1936 + Hc * 44 + Wc) * C + n] =
              acc[mt][j] + bias;
        }
      }
    }
  }
}

// ------------------------------------------------------------------
extern "C" void kernel_launch(void* const* d_in, const int* in_sizes, int n_in,
                              void* d_out, int out_size, void* d_ws, size_t ws_size,
                              hipStream_t stream) {
  const float* x      = (const float*)d_in[0];
  const float* qkv_w  = (const float*)d_in[1];
  const float* proj_w = (const float*)d_in[2];
  const float* proj_b = (const float*)d_in[3];
  // d_in[4..6] = t, H, W (fixed at 4, 44, 44 for this instance)

  const size_t Exw = (size_t)NWIN * TOK * C;        // 7.68M bf16
  const size_t Eh  = (size_t)NWIN * NH * TOK * HD;  // 7.68M bf16
  __bf16* xw = (__bf16*)d_ws;
  __bf16* q  = xw + Exw;
  __bf16* kk = q + Eh;
  __bf16* vt = kk + Eh;
  __bf16* ow = vt + Eh;
  __bf16* wq = ow + Exw;                            // 1152*384 bf16
  __bf16* wp = wq + (size_t)C3 * C;                 // 384*384 bf16
  float* out = (float*)d_out;

  int gatherBlocks = (NWIN * TOK * (C / 4) + 255) / 256;   // 7500
  gather_win <<<gatherBlocks, 256, 0, stream>>>(x, xw);
  cvt_weights<<<(C3 * C / 4 + 255) / 256, 256, 0, stream>>>(qkv_w, wq, C3 * C / 4);
  cvt_weights<<<(C * C / 4 + 255) / 256, 256, 0, stream>>>(proj_w, wp, C * C / 4);
  qkv_gemm   <<<NWIN * 5, 256, 0, stream>>>(xw, wq, q, kk, vt);
  attn_kernel<<<NWIN * NH, 256, 0, stream>>>(q, kk, vt, ow);
  proj_gemm  <<<NWIN * 5, 256, 0, stream>>>(ow, wp, proj_b, out);
}
